// SwinV23DLayer_39505109189141
// MI455X (gfx1250) — compile-verified, matrix operand order checked by reading, not run on hardware
//
#include <hip/hip_runtime.h>

typedef __attribute__((ext_vector_type(16))) _Float16 v16h;
typedef __attribute__((ext_vector_type(8)))  _Float16 v8h;
typedef __attribute__((ext_vector_type(8)))  float    v8f;

#define WMMA_F16(a, b, c) \
  __builtin_amdgcn_wmma_f32_16x16x32_f16(false, (a), false, (b), (short)0, (c), false, false)

// ---------------------------------------------------------------------------
// Fragment loaders for V_WMMA_F32_16X16X32_F16 (wave32).
// 16-bit A/B layout (ISA 7.12.2): lanes 0-15 -> rows 0-15, K = {0..7, 16..23};
// lanes 16-31 -> rows 0-15, K = {8..15, 24..31}.
// ---------------------------------------------------------------------------
__device__ inline v16h load_frag_k1(const _Float16* p0, int mstride) {
  int lane = threadIdx.x & 31;
  const _Float16* p = p0 + (lane & 15) * mstride + ((lane >> 4) << 3);
  v8h lo = *(const v8h*)(p);
  v8h hi = *(const v8h*)(p + 16);
  v16h r;
#pragma unroll
  for (int i = 0; i < 8; ++i) { r[i] = lo[i]; r[8 + i] = hi[i]; }
  return r;
}

__device__ inline v16h load_frag_strided(const _Float16* p0, int nstride, int kstride) {
  int lane = threadIdx.x & 31;
  const _Float16* p = p0 + (lane & 15) * nstride + ((lane >> 4) << 3) * kstride;
  v16h r;
#pragma unroll
  for (int i = 0; i < 8; ++i) { r[i] = p[i * kstride]; r[8 + i] = p[(16 + i) * kstride]; }
  return r;
}

// ---------------------------------------------------------------------------
// Weight prep: fp32 [K][N] -> f16 transposed [N][K] in workspace.
// Sizes: qkv 288x96, proj 96x96, fc1 384x96, fc2 96x384 (110592 elems total).
// ---------------------------------------------------------------------------
__global__ __launch_bounds__(256) void prep_weights(
    const float* __restrict__ wqkv, const float* __restrict__ wproj,
    const float* __restrict__ wfc1, const float* __restrict__ wfc2,
    _Float16* __restrict__ wqkvT, _Float16* __restrict__ wprojT,
    _Float16* __restrict__ wfc1T, _Float16* __restrict__ wfc2T) {
  int i = blockIdx.x * 256 + threadIdx.x;
  if (i < 27648) {                      // qkvT [288][96]
    int n = i / 96, k = i % 96;
    wqkvT[i] = (_Float16)wqkv[k * 288 + n];
  } else if (i < 36864) {               // projT [96][96]
    int j = i - 27648; int n = j / 96, k = j % 96;
    wprojT[j] = (_Float16)wproj[k * 96 + n];
  } else if (i < 73728) {               // fc1T [384][96]
    int j = i - 36864; int n = j / 96, k = j % 96;
    wfc1T[j] = (_Float16)wfc1[k * 384 + n];
  } else if (i < 110592) {              // fc2T [96][384]
    int j = i - 73728; int n = j / 384, k = j % 384;
    wfc2T[j] = (_Float16)wfc2[k * 96 + n];
  }
}

// ---------------------------------------------------------------------------
// CPB meta-network table: tab[t][h] = 16*sigmoid(relu(coords(t)@w1+b1)@w2),
// t in [0,343), coords = log-spaced SwinV2 relative coords for window 4x4x4.
// ---------------------------------------------------------------------------
__global__ __launch_bounds__(64) void cpb_table(
    const float* __restrict__ w1, const float* __restrict__ b1,
    const float* __restrict__ w2, float* __restrict__ tab) {
  __shared__ float red[64 * 3];
  int t = blockIdx.x;                               // 0..342
  int iz = t / 49, rem = t % 49, iy = rem / 7, ix = rem % 7;
  float rv[3] = {(float)(iz - 3), (float)(iy - 3), (float)(ix - 3)};
  float c[3];
#pragma unroll
  for (int d = 0; d < 3; ++d) {
    float v = rv[d] * (8.0f / 3.0f);
    float s = (v > 0.f) ? 1.f : ((v < 0.f) ? -1.f : 0.f);
    c[d] = s * log2f(fabsf(v) + 1.0f) * (1.0f / 3.0f);   // /log2(8)
  }
  int j = threadIdx.x;
  float a0 = 0.f, a1 = 0.f, a2 = 0.f;
  for (int h = j; h < 512; h += 64) {
    float hv = c[0] * w1[h] + c[1] * w1[512 + h] + c[2] * w1[1024 + h] + b1[h];
    hv = fmaxf(hv, 0.f);
    a0 += hv * w2[h * 3 + 0];
    a1 += hv * w2[h * 3 + 1];
    a2 += hv * w2[h * 3 + 2];
  }
  red[j * 3 + 0] = a0; red[j * 3 + 1] = a1; red[j * 3 + 2] = a2;
  __syncthreads();
  if (j < 3) {
    float s = 0.f;
    for (int q = 0; q < 64; ++q) s += red[q * 3 + j];
    tab[t * 3 + j] = 16.0f / (1.0f + __expf(-s));
  }
}

// ---------------------------------------------------------------------------
// Main per-window kernel. One block = one 4x4x4 window (N=64 tokens, C=96).
// LDS layout (159744 B total, dynamic; dynamic LDS starts at wave LDS offset 0):
//   XRES  f32 [64][96]        @      0  (24576)   <- filled by async DMA
//   XH    f16 [64][96]        @  24576  (12288)
//   QKF   f32 [64][192]       @  36864  (49152)  } aliased: LOGITS f32[3][64][64],
//                                                 }          H1H f16[64][384]
//   QH    f16 [3][64][32]     @  86016  (12288)  } aliased by PH f16[3][64][64]
//   KH    f16 [3][64][32]     @  98304  (12288)  }
//   VH    f16 [3][64][32]     @ 110592  (12288)
//   AOH   f16 [64][96]        @ 122880  (12288)
//   TMPF  f32 [64][96]        @ 135168  (24576)
// ---------------------------------------------------------------------------
__global__ __launch_bounds__(256) void swin_window_kernel(
    const float* __restrict__ x_in,
    const float* __restrict__ b_qkv, const float* __restrict__ b_proj,
    const float* __restrict__ logit_scale,
    const float* __restrict__ ln1_g, const float* __restrict__ ln1_b,
    const float* __restrict__ b_fc1, const float* __restrict__ b_fc2,
    const float* __restrict__ ln2_g, const float* __restrict__ ln2_b,
    const float* __restrict__ tab,
    const _Float16* __restrict__ wqkvT, const _Float16* __restrict__ wprojT,
    const _Float16* __restrict__ wfc1T, const _Float16* __restrict__ wfc2T,
    float* __restrict__ out) {
  extern __shared__ char smem[];
  float*    XRES   = (float*)(smem);
  _Float16* XH     = (_Float16*)(smem + 24576);
  float*    QKF    = (float*)(smem + 36864);
  float*    LOGITS = (float*)(smem + 36864);
  _Float16* H1H    = (_Float16*)(smem + 36864);
  _Float16* QH     = (_Float16*)(smem + 86016);
  _Float16* KH     = (_Float16*)(smem + 98304);
  _Float16* PH     = (_Float16*)(smem + 86016);
  _Float16* VH     = (_Float16*)(smem + 110592);
  _Float16* AOH    = (_Float16*)(smem + 122880);
  float*    TMPF   = (float*)(smem + 135168);

  const int tid  = threadIdx.x;
  const int lane = tid & 31;
  const int wave = tid >> 5;

  // window index -> (b, iz, iy, ix); token n -> (zz,yy,xx) = (n>>4,(n>>2)&3,n&3)
  int w  = blockIdx.x;
  int ix = w & 15, iy = (w >> 4) & 15, iz = (w >> 8) & 7, bb = w >> 11;

  // ---- stage 0a: async DMA of the window tile into LDS (global -> XRES) ----
  // Each token row is 384 contiguous bytes in global; 64 rows x 24 16B chunks
  // = 1536 b128 async transfers (6 per thread), tracked on ASYNCcnt.
  for (int e = tid; e < 1536; e += 256) {
    int n = e / 24, ch = e % 24;
    int zz = n >> 4, yy = (n >> 2) & 3, xx = n & 3;
    size_t g = ((((size_t)(bb * 32 + iz * 4 + zz)) * 64 + (iy * 4 + yy)) * 64 +
                (ix * 4 + xx)) * 96 + ch * 4;
    const float* gsrc = x_in + g;
    unsigned ldsoff = (unsigned)(n * 384 + ch * 16);   // XRES @ LDS offset 0
    asm volatile("global_load_async_to_lds_b128 %0, %1, off"
                 :: "v"(ldsoff), "v"(gsrc) : "memory");
  }
  // Warm L2/WGP$ for the f16 weights while the DMA is in flight.
  if (tid == 0) {
    __builtin_prefetch(wqkvT, 0, 0);
    __builtin_prefetch(wprojT, 0, 0);
    __builtin_prefetch(wfc1T, 0, 0);
    __builtin_prefetch(wfc2T, 0, 0);
    __builtin_prefetch(tab, 0, 0);
  }
  asm volatile("s_wait_asynccnt 0x0" ::: "memory");
  __syncthreads();

  // ---- stage 0b: f32 -> f16 operand copy ----
  for (int e = tid; e < 64 * 96; e += 256) XH[e] = (_Float16)XRES[e];
  __syncthreads();

  // ---- stage 1: QKV GEMM [64,96]x[96,288], q/k -> f32 LDS, v -> f16 ----
  for (int t = wave; t < 72; t += 8) {
    int mt = t / 18, nt = t % 18;
    v8f acc = {};
#pragma unroll
    for (int kt = 0; kt < 3; ++kt) {
      v16h a  = load_frag_k1(XH + (mt * 16) * 96 + kt * 32, 96);
      v16h bf = load_frag_k1(wqkvT + (nt * 16) * 96 + kt * 32, 96);
      acc = WMMA_F16(a, bf, acc);
    }
    int nn = lane & 15, m0 = (lane >> 4) * 8;
    int col = nt * 16 + nn;
    float bias = b_qkv[col];
    if (nt < 12) {                       // q,k tiles -> f32 (uniform branch)
#pragma unroll
      for (int r = 0; r < 8; ++r)
        QKF[(mt * 16 + m0 + r) * 192 + col] = acc[r] + bias;
    } else {                             // v tiles -> f16 head layout
      int cc = col - 192, h = cc >> 5, d = cc & 31;
#pragma unroll
      for (int r = 0; r < 8; ++r)
        VH[(h * 64 + mt * 16 + m0 + r) * 32 + d] = (_Float16)(acc[r] + bias);
    }
  }
  __syncthreads();

  // ---- stage 2: cosine-normalize q and k rows -> f16 ----
  for (int e = tid; e < 384; e += 256) {
    int isK = (e >= 192) ? 1 : 0;
    int r = e - 192 * isK, h = r / 64, m = r % 64;
    int col0 = 96 * isK + h * 32;
    float ss = 0.f;
    for (int d = 0; d < 32; ++d) { float v = QKF[m * 192 + col0 + d]; ss += v * v; }
    float inv = 1.0f / fmaxf(sqrtf(ss), 1e-12f);
    _Float16* dst = (isK ? KH : QH) + (h * 64 + m) * 32;
    for (int d = 0; d < 32; ++d) dst[d] = (_Float16)(QKF[m * 192 + col0 + d] * inv);
  }
  __syncthreads();

  // ---- stage 3: logits = (q.k^T)*exp(min(ls,log100)) + cpb_bias ----
  for (int t = wave; t < 48; t += 8) {
    int h = t / 16, tt = t % 16, mt = tt / 4, nt = tt % 4;
    v16h a  = load_frag_k1(QH + (h * 64 + mt * 16) * 32, 32);
    v16h bf = load_frag_k1(KH + (h * 64 + nt * 16) * 32, 32);
    v8f acc = {};
    acc = WMMA_F16(a, bf, acc);
    float sc = __expf(fminf(logit_scale[h], 4.6051702f));
    int nn = lane & 15, m0 = (lane >> 4) * 8;
    int nk = nt * 16 + nn;
    int zk = nk >> 4, yk = (nk >> 2) & 3, xk = nk & 3;
#pragma unroll
    for (int r = 0; r < 8; ++r) {
      int mq = mt * 16 + m0 + r;
      int zq = mq >> 4, yq = (mq >> 2) & 3, xq = mq & 3;
      int idx = (zq - zk + 3) * 49 + (yq - yk + 3) * 7 + (xq - xk + 3);
      LOGITS[(h * 64 + mq) * 64 + nk] = acc[r] * sc + tab[idx * 3 + h];
    }
  }
  __syncthreads();

  // ---- stage 4: softmax rows -> f16 probs (PH aliases QH/KH) ----
  for (int e = tid; e < 192; e += 256) {
    const float* row = LOGITS + e * 64;
    float mx = row[0];
    for (int n = 1; n < 64; ++n) mx = fmaxf(mx, row[n]);
    float s = 0.f;
    for (int n = 0; n < 64; ++n) s += __expf(row[n] - mx);
    float invs = 1.0f / s;
    _Float16* prow = PH + e * 64;
    for (int n = 0; n < 64; ++n) prow[n] = (_Float16)(__expf(row[n] - mx) * invs);
  }
  __syncthreads();

  // ---- stage 5: out_h = probs @ v  (per head, [64,64]x[64,32]) ----
  for (int t = wave; t < 24; t += 8) {
    int h = t / 8, tt = t % 8, mt = tt / 2, nt = tt % 2;
    v8f acc = {};
#pragma unroll
    for (int kt = 0; kt < 2; ++kt) {
      v16h a  = load_frag_k1(PH + (h * 64 + mt * 16) * 64 + kt * 32, 64);
      v16h bf = load_frag_strided(VH + h * 2048 + (kt * 32) * 32 + nt * 16, 1, 32);
      acc = WMMA_F16(a, bf, acc);
    }
    int nn = lane & 15, m0 = (lane >> 4) * 8;
#pragma unroll
    for (int r = 0; r < 8; ++r)
      AOH[(mt * 16 + m0 + r) * 96 + h * 32 + nt * 16 + nn] = (_Float16)acc[r];
  }
  __syncthreads();

  // ---- stage 6: proj GEMM [64,96]x[96,96] -> TMPF ----
  for (int t = wave; t < 24; t += 8) {
    int mt = t / 6, nt = t % 6;
    v8f acc = {};
#pragma unroll
    for (int kt = 0; kt < 3; ++kt) {
      v16h a  = load_frag_k1(AOH + (mt * 16) * 96 + kt * 32, 96);
      v16h bf = load_frag_k1(wprojT + (nt * 16) * 96 + kt * 32, 96);
      acc = WMMA_F16(a, bf, acc);
    }
    int nn = lane & 15, m0 = (lane >> 4) * 8;
    int col = nt * 16 + nn;
    float bias = b_proj[col];
#pragma unroll
    for (int r = 0; r < 8; ++r) TMPF[(mt * 16 + m0 + r) * 96 + col] = acc[r] + bias;
  }
  __syncthreads();

  // ---- stage 7: x = x + LN1(proj_out)  (residual-post-norm) ----
  if (tid < 64) {
    const float* row = TMPF + tid * 96;
    float mu = 0.f;
    for (int c = 0; c < 96; ++c) mu += row[c];
    mu *= (1.0f / 96.0f);
    float var = 0.f;
    for (int c = 0; c < 96; ++c) { float d = row[c] - mu; var += d * d; }
    var *= (1.0f / 96.0f);
    float rs = rsqrtf(var + 1e-5f);
    for (int c = 0; c < 96; ++c) {
      float v  = (row[c] - mu) * rs * ln1_g[c] + ln1_b[c];
      float nx = XRES[tid * 96 + c] + v;
      XRES[tid * 96 + c] = nx;
      XH[tid * 96 + c] = (_Float16)nx;
    }
  }
  __syncthreads();

  // ---- stage 8: fc1 GEMM [64,96]x[96,384] + GELU(tanh) -> H1H f16 ----
  for (int t = wave; t < 96; t += 8) {
    int mt = t / 24, nt = t % 24;
    v8f acc = {};
#pragma unroll
    for (int kt = 0; kt < 3; ++kt) {
      v16h a  = load_frag_k1(XH + (mt * 16) * 96 + kt * 32, 96);
      v16h bf = load_frag_k1(wfc1T + (nt * 16) * 96 + kt * 32, 96);
      acc = WMMA_F16(a, bf, acc);
    }
    int nn = lane & 15, m0 = (lane >> 4) * 8;
    int col = nt * 16 + nn;
    float bias = b_fc1[col];
#pragma unroll
    for (int r = 0; r < 8; ++r) {
      float xv = acc[r] + bias;
      float g = 0.5f * xv * (1.0f + tanhf(0.7978845608f * (xv + 0.044715f * xv * xv * xv)));
      H1H[(mt * 16 + m0 + r) * 384 + col] = (_Float16)g;
    }
  }
  __syncthreads();

  // ---- stage 9: fc2 GEMM [64,384]x[384,96] -> TMPF ----
  for (int t = wave; t < 24; t += 8) {
    int mt = t / 6, nt = t % 6;
    v8f acc = {};
    for (int kt = 0; kt < 12; ++kt) {
      v16h a  = load_frag_k1(H1H + (mt * 16) * 384 + kt * 32, 384);
      v16h bf = load_frag_k1(wfc2T + (nt * 16) * 384 + kt * 32, 384);
      acc = WMMA_F16(a, bf, acc);
    }
    int nn = lane & 15, m0 = (lane >> 4) * 8;
    int col = nt * 16 + nn;
    float bias = b_fc2[col];
#pragma unroll
    for (int r = 0; r < 8; ++r) TMPF[(mt * 16 + m0 + r) * 96 + col] = acc[r] + bias;
  }
  __syncthreads();

  // ---- stage 10a: final = x + LN2(mlp_out), written back into TMPF ----
  if (tid < 64) {
    const float* row = TMPF + tid * 96;
    float mu = 0.f;
    for (int c = 0; c < 96; ++c) mu += row[c];
    mu *= (1.0f / 96.0f);
    float var = 0.f;
    for (int c = 0; c < 96; ++c) { float d = row[c] - mu; var += d * d; }
    var *= (1.0f / 96.0f);
    float rs = rsqrtf(var + 1e-5f);
    for (int c = 0; c < 96; ++c) {
      float v = (row[c] - mu) * rs * ln2_g[c] + ln2_b[c];
      TMPF[tid * 96 + c] = XRES[tid * 96 + c] + v;
    }
  }
  __syncthreads();

  // ---- stage 10b: scatter window back to global (window merge) ----
  for (int e = tid; e < 64 * 96; e += 256) {
    int n = e / 96, c = e % 96;
    int zz = n >> 4, yy = (n >> 2) & 3, xx = n & 3;
    size_t g = ((((size_t)(bb * 32 + iz * 4 + zz)) * 64 + (iy * 4 + yy)) * 64 +
                (ix * 4 + xx)) * 96 + c;
    out[g] = TMPF[n * 96 + c];
  }
}

// ---------------------------------------------------------------------------
// Launch. Workspace layout (bytes): tab @0 (4116), wqkvT @4608 (55296),
// wprojT @59904 (18432), wfc1T @78336 (73728), wfc2T @152064 (73728).
// ---------------------------------------------------------------------------
extern "C" void kernel_launch(void* const* d_in, const int* in_sizes, int n_in,
                              void* d_out, int out_size, void* d_ws, size_t ws_size,
                              hipStream_t stream) {
  (void)in_sizes; (void)n_in; (void)out_size; (void)ws_size;
  const float* x      = (const float*)d_in[0];
  const float* wqkv   = (const float*)d_in[1];
  const float* bqkv   = (const float*)d_in[2];
  const float* wproj  = (const float*)d_in[3];
  const float* bproj  = (const float*)d_in[4];
  const float* lscale = (const float*)d_in[5];
  const float* cw1    = (const float*)d_in[6];
  const float* cb1    = (const float*)d_in[7];
  const float* cw2    = (const float*)d_in[8];
  const float* l1g    = (const float*)d_in[9];
  const float* l1b    = (const float*)d_in[10];
  const float* wfc1   = (const float*)d_in[11];
  const float* bfc1   = (const float*)d_in[12];
  const float* wfc2   = (const float*)d_in[13];
  const float* bfc2   = (const float*)d_in[14];
  const float* l2g    = (const float*)d_in[15];
  const float* l2b    = (const float*)d_in[16];
  float* out = (float*)d_out;

  char* ws = (char*)d_ws;
  float*    tab    = (float*)(ws + 0);
  _Float16* wqkvT  = (_Float16*)(ws + 4608);
  _Float16* wprojT = (_Float16*)(ws + 59904);
  _Float16* wfc1T  = (_Float16*)(ws + 78336);
  _Float16* wfc2T  = (_Float16*)(ws + 152064);

  prep_weights<<<432, 256, 0, stream>>>(wqkv, wproj, wfc1, wfc2,
                                        wqkvT, wprojT, wfc1T, wfc2T);
  cpb_table<<<343, 64, 0, stream>>>(cw1, cb1, cw2, tab);

  (void)hipFuncSetAttribute(reinterpret_cast<const void*>(swin_window_kernel),
                            hipFuncAttributeMaxDynamicSharedMemorySize, 160 * 1024);
  swin_window_kernel<<<4096, 256, 159744, stream>>>(
      x, bqkv, bproj, lscale, l1g, l1b, bfc1, bfc2, l2g, l2b,
      tab, wqkvT, wprojT, wfc1T, wfc2T, out);
}
